// SDFBlock_45578192945779
// MI455X (gfx1250) — matrix operand order repaired and checked by reading, not compiled, not run on hardware
//
#include <hip/hip_runtime.h>
#include <math.h>

typedef __attribute__((ext_vector_type(16))) __bf16 v16bf;
typedef __attribute__((ext_vector_type(8)))  float  v8f;
typedef __attribute__((ext_vector_type(2)))  float  v2f;

#define B_      8
#define C_      192
#define HEADS_  6
#define HD_     32
#define NTOK    100352          /* B*112*112 */
#define NWIN    2048            /* B*16*16   */
#define KF_     57              /* rfft width */
#define NFREQ   51072           /* B*112*57  */
#define SCALE_  0.17677669529663687f
#define LNEPS   1e-5f

__device__ __forceinline__ float gelu_exact(float x) {
    return 0.5f * x * (1.0f + erff(x * 0.70710678118654752f));
}

/* low 32 bits of a generic pointer into LDS == LDS byte offset (ISA aperture rule) */
__device__ __forceinline__ unsigned lds_off(const void* p) {
    return (unsigned)(unsigned long long)p;
}

/* async global->LDS copy of 16 bytes per lane (tracked by ASYNCcnt) */
__device__ __forceinline__ void async_copy_b128(unsigned lds_dst, const void* gsrc) {
    unsigned long long ga = (unsigned long long)gsrc;
    asm volatile("global_load_async_to_lds_b128 %0, %1, off"
                 :: "v"(lds_dst), "v"(ga) : "memory");
}
__device__ __forceinline__ void wait_asynccnt0() {
    asm volatile("s_wait_asynccnt 0x0" ::: "memory");
}

/* ---------------- f32 -> bf16 conversion (plain) ---------------- */
__global__ void f2bf_kernel(const float* __restrict__ s, __bf16* __restrict__ d, int n) {
    int i = blockIdx.x * blockDim.x + threadIdx.x;
    if (i < n) d[i] = (__bf16)s[i];
}

/* ---------------- f32 KxN -> bf16 NxK (transpose at convert time) ---------------- */
__global__ void f2bft_kernel(const float* __restrict__ s, __bf16* __restrict__ d, int K, int N) {
    int i = blockIdx.x * blockDim.x + threadIdx.x;
    if (i < K * N) {
        int k = i / N, n = i % N;
        d[(size_t)n * K + k] = (__bf16)s[i];
    }
}

/* ---------------- twiddle table: cos/sin(2*pi*j/112) ---------------- */
__global__ void twiddle_init_kernel(float* __restrict__ tw) {
    int j = threadIdx.x;
    if (j < 112) {
        float ang = 6.283185307179586f * (float)j / 112.0f;
        tw[j]       = cosf(ang);
        tw[112 + j] = sinf(ang);
    }
}

/* ---------------- attention position-bias tables ---------------- */
__global__ void biastab_kernel(const float* __restrict__ an_b, const float* __restrict__ na_b,
                               const float* __restrict__ ah_b, const float* __restrict__ aw_b,
                               const float* __restrict__ ha_b, const float* __restrict__ wa_b,
                               float* __restrict__ pb, float* __restrict__ ab) {
    int idx = blockIdx.x * blockDim.x + threadIdx.x;
    if (idx >= HEADS_ * 49 * 49) return;
    int h = idx / 2401, rem = idx % 2401;
    int a = rem / 49, n = rem % 49;
    int iy = n / 7, ix = n % 7;
    pb[idx] = an_b[idx] + ah_b[(h * 49 + a) * 7 + iy] + aw_b[(h * 49 + a) * 7 + ix];
    int nn = rem / 49, aa = rem % 49;
    int jy = nn / 7, jx = nn % 7;
    ab[(h * 49 + nn) * 49 + aa] =
        na_b[(h * 49 + aa) * 49 + nn] + ha_b[h * 343 + jy * 49 + aa] + wa_b[h * 343 + jx * 49 + aa];
}

/* ---------------- LN1 + window partition + bf16 pack ---------------- */
__global__ __launch_bounds__(192)
void prep_kernel(const float* __restrict__ xF1, const float* __restrict__ xC2,
                 const float* __restrict__ xC1, const float* __restrict__ g1,
                 const float* __restrict__ b1,
                 __bf16* __restrict__ qw, __bf16* __restrict__ kin, __bf16* __restrict__ vin,
                 __bf16* __restrict__ c1w, __bf16* __restrict__ c2w) {
    __shared__ float red[256];
    __shared__ float s_mean, s_rstd;
    int t = blockIdx.x, c = threadIdx.x;
    size_t base = (size_t)t * C_ + c;
    float x = xF1[base];
    red[c] = x; if (c < 64) red[192 + c] = 0.f;
    __syncthreads();
    for (int s = 128; s > 0; s >>= 1) { if (c < s) red[c] += red[c + s]; __syncthreads(); }
    if (c == 0) s_mean = red[0] * (1.f / 192.f);
    __syncthreads();
    float m = s_mean, d = x - m;
    red[c] = d * d; if (c < 64) red[192 + c] = 0.f;
    __syncthreads();
    for (int s = 128; s > 0; s >>= 1) { if (c < s) red[c] += red[c + s]; __syncthreads(); }
    if (c == 0) s_rstd = rsqrtf(red[0] * (1.f / 192.f) + LNEPS);
    __syncthreads();
    float xn = d * s_rstd * g1[c] + b1[c];
    int b = t / 12544, n = t % 12544;
    int hh = n / 112, ww = n % 112;
    int m_row = (((b * 16 + hh / 7) * 16 + ww / 7) * 49) + (hh % 7) * 7 + (ww % 7);
    size_t o = (size_t)m_row * C_ + c;
    float v2 = xC2[base], v1 = xC1[base];
    qw[o]  = (__bf16)xn;
    kin[o] = (__bf16)(xn + v2 - v1);
    vin[o] = (__bf16)(v2 - v1);
    c1w[o] = (__bf16)v1;
    c2w[o] = (__bf16)v2;
}

/* ---------------- bf16 WMMA GEMM: C = A(MxK) @ Bt(NxK)^T + bias (+gelu) ----------------
 * Bt is the weight pre-transposed to N-major at convert time, so BOTH tiles stage via
 * async global->LDS b128 copies (ASYNCcnt), double-buffered, 1 barrier per K-chunk.
 * Requires M%128==0, N%64==0, K%32==0 (true for all call sites here). */
#define GBM 128
#define GBN 64
#define GBK 32
#define ALD 40
#define BLD 40
__global__ __launch_bounds__(256)
void gemm_bf16_kernel(const __bf16* __restrict__ A, const __bf16* __restrict__ Bt,
                      const float* __restrict__ bias, float* __restrict__ outF,
                      __bf16* __restrict__ outH, int M, int N, int K, int doGelu) {
    __shared__ __bf16 As[2][GBM * ALD];
    __shared__ __bf16 Bs[2][GBN * BLD];
    int tid = threadIdx.x;
    int wave = tid >> 5, lane = tid & 31;
    int wm = wave & 3, wn = wave >> 2;
    int r = lane & 15, hi = lane >> 4;
    int m_blk = blockIdx.y * GBM;
    int n_blk = blockIdx.x * GBN;

    v8f acc[2][2];
#pragma unroll
    for (int i = 0; i < 2; ++i)
#pragma unroll
        for (int j = 0; j < 2; ++j)
#pragma unroll
            for (int e = 0; e < 8; ++e) acc[i][j][e] = 0.f;

    auto stageA = [&](int k0, int buf) {
#pragma unroll
        for (int it = 0; it < 2; ++it) {
            int idx = (tid + it * 256) * 8;      /* 8 bf16 = 16B per lane */
            int row = idx >> 5, col = idx & 31;
            const __bf16* src = A + (size_t)(m_blk + row) * K + (k0 + col);
            async_copy_b128(lds_off(&As[buf][row * ALD + col]), src);
        }
    };
    auto stageB = [&](int k0, int buf) {
        int idx = tid * 8;
        int n0 = idx >> 5, kcol = idx & 31;
        const __bf16* src = Bt + (size_t)(n_blk + n0) * K + (k0 + kcol);
        async_copy_b128(lds_off(&Bs[buf][n0 * BLD + kcol]), src);
    };

    stageA(0, 0);
    stageB(0, 0);
    wait_asynccnt0();
    __syncthreads();

    const int nk = K / GBK;
    for (int kc = 0; kc < nk; ++kc) {
        int buf = kc & 1;
        if (kc + 1 < nk) {                      /* overlap next-tile DMA with WMMA */
            stageA((kc + 1) * GBK, buf ^ 1);
            stageB((kc + 1) * GBK, buf ^ 1);
        }
        v16bf af[2], bfv[2];
#pragma unroll
        for (int ti = 0; ti < 2; ++ti) {
            const __bf16* basep = &As[buf][(wm * 32 + ti * 16 + r) * ALD];
#pragma unroll
            for (int e = 0; e < 8; ++e) {
                af[ti][e]     = basep[hi * 8 + e];
                af[ti][8 + e] = basep[16 + hi * 8 + e];
            }
        }
#pragma unroll
        for (int tj = 0; tj < 2; ++tj) {
            const __bf16* basep = &Bs[buf][(wn * 32 + tj * 16 + r) * BLD];
#pragma unroll
            for (int e = 0; e < 8; ++e) {
                bfv[tj][e]     = basep[hi * 8 + e];
                bfv[tj][8 + e] = basep[16 + hi * 8 + e];
            }
        }
#pragma unroll
        for (int ti = 0; ti < 2; ++ti)
#pragma unroll
            for (int tj = 0; tj < 2; ++tj)
                acc[ti][tj] = __builtin_amdgcn_wmma_f32_16x16x32_bf16(
                    false, af[ti], false, bfv[tj], (short)0, acc[ti][tj], false, false);
        if (kc + 1 < nk) {
            wait_asynccnt0();                   /* next tiles resident in LDS */
            __syncthreads();
        }
    }

    /* epilogue: compute once, then hoisted store paths */
    float vals[2][2][8];
#pragma unroll
    for (int ti = 0; ti < 2; ++ti)
#pragma unroll
        for (int tj = 0; tj < 2; ++tj) {
            int col = n_blk + wn * 32 + tj * 16 + r;
            float bval = bias ? bias[col] : 0.f;
#pragma unroll
            for (int e = 0; e < 8; ++e) {
                float val = acc[ti][tj][e] + bval;
                if (doGelu) val = gelu_exact(val);
                vals[ti][tj][e] = val;
            }
        }
    if (outF) {
#pragma unroll
        for (int ti = 0; ti < 2; ++ti)
#pragma unroll
            for (int tj = 0; tj < 2; ++tj) {
                int col = n_blk + wn * 32 + tj * 16 + r;
#pragma unroll
                for (int e = 0; e < 8; ++e) {
                    int row = m_blk + wm * 32 + ti * 16 + hi * 8 + e;
                    outF[(size_t)row * N + col] = vals[ti][tj][e];
                }
            }
    }
    if (outH) {
#pragma unroll
        for (int ti = 0; ti < 2; ++ti)
#pragma unroll
            for (int tj = 0; tj < 2; ++tj) {
                int col = n_blk + wn * 32 + tj * 16 + r;
#pragma unroll
                for (int e = 0; e < 8; ++e) {
                    int row = m_blk + wm * 32 + ti * 16 + hi * 8 + e;
                    outH[(size_t)row * N + col] = (__bf16)vals[ti][tj][e];
                }
            }
    }
}

/* ---------------- f32 WMMA GEMM pair: C = A1@B1 + sign2*(A2@B2) (+gelu) ----------------
 * Full-precision path for the spectral (complex) MLP via v_wmma_f32_16x16x4_f32. */
#define FLD 20
__global__ __launch_bounds__(256)
void gemm_f32x2_kernel(const float* __restrict__ A1, const float* __restrict__ B1,
                       const float* __restrict__ A2, const float* __restrict__ B2,
                       float sign2, float* __restrict__ outF,
                       int M, int N, int K, int doGelu) {
    __shared__ float As[2][128 * FLD];
    __shared__ float Bs[2][64 * FLD];
    int tid = threadIdx.x;
    int wave = tid >> 5, lane = tid & 31;
    int wm = wave & 3, wn = wave >> 2;
    int r = lane & 15, hi = lane >> 4;
    int m_blk = blockIdx.y * 128, n_blk = blockIdx.x * 64;

    v8f acc[2][2];
#pragma unroll
    for (int i = 0; i < 2; ++i)
#pragma unroll
        for (int j = 0; j < 2; ++j)
#pragma unroll
            for (int e = 0; e < 8; ++e) acc[i][j][e] = 0.f;

    for (int k0 = 0; k0 < K; k0 += 16) {
        {
            int idx = tid * 8;
            int row = idx >> 4, col = idx & 15;
            const float* s1 = A1 + (size_t)(m_blk + row) * K + k0 + col;
            const float* s2 = A2 + (size_t)(m_blk + row) * K + k0 + col;
#pragma unroll
            for (int e = 0; e < 8; ++e) {
                As[0][row * FLD + col + e] = s1[e];
                As[1][row * FLD + col + e] = s2[e] * sign2;
            }
        }
        {
            int idx = tid * 4;
            int kk = idx >> 6, n0 = idx & 63;
            const float* s1 = B1 + (size_t)(k0 + kk) * N + n_blk + n0;
            const float* s2 = B2 + (size_t)(k0 + kk) * N + n_blk + n0;
#pragma unroll
            for (int e = 0; e < 4; ++e) {
                Bs[0][(n0 + e) * FLD + kk] = s1[e];
                Bs[1][(n0 + e) * FLD + kk] = s2[e];
            }
        }
        if (k0 + 16 < K) {                       /* near-cache prefetch of next A1 rows */
            __builtin_prefetch(A1 + (size_t)(m_blk + (tid >> 1)) * K + (k0 + 16), 0, 3);
        }
        __syncthreads();
#pragma unroll
        for (int src = 0; src < 2; ++src) {
#pragma unroll
            for (int ks = 0; ks < 16; ks += 4) {
                v2f af[2], bfv[2];
#pragma unroll
                for (int ti = 0; ti < 2; ++ti) {
                    const float* basep = &As[src][(wm * 32 + ti * 16 + r) * FLD + ks + hi * 2];
                    af[ti][0] = basep[0]; af[ti][1] = basep[1];
                }
#pragma unroll
                for (int tj = 0; tj < 2; ++tj) {
                    const float* basep = &Bs[src][(wn * 32 + tj * 16 + r) * FLD + ks + hi * 2];
                    bfv[tj][0] = basep[0]; bfv[tj][1] = basep[1];
                }
#pragma unroll
                for (int ti = 0; ti < 2; ++ti)
#pragma unroll
                    for (int tj = 0; tj < 2; ++tj)
                        acc[ti][tj] = __builtin_amdgcn_wmma_f32_16x16x4_f32(
                            false, af[ti], false, bfv[tj], (short)0, acc[ti][tj], false, false);
            }
        }
        __syncthreads();
    }
#pragma unroll
    for (int ti = 0; ti < 2; ++ti)
#pragma unroll
        for (int tj = 0; tj < 2; ++tj) {
            int col = n_blk + wn * 32 + tj * 16 + r;
#pragma unroll
            for (int e = 0; e < 8; ++e) {
                int row = m_blk + wm * 32 + ti * 16 + hi * 8 + e;
                float val = acc[ti][tj][e];
                if (doGelu) val = gelu_exact(val);
                outF[(size_t)row * N + col] = val;
            }
        }
}

/* ---------------- agent attention (per window, per head) ---------------- */
__global__ __launch_bounds__(128)
void attn_kernel(const float* __restrict__ qf, const float* __restrict__ kf,
                 const float* __restrict__ vf, const float* __restrict__ a1f,
                 const float* __restrict__ a2f, const float* __restrict__ pb,
                 const float* __restrict__ ab, float* __restrict__ xw) {
    __shared__ float sq[49][32], sk[49][32], sv[49][32], sa1[49][32], sa2[49][32];
    __shared__ float S[49][52];
    __shared__ float av[49][32];
    int win = blockIdx.x / HEADS_, head = blockIdx.x % HEADS_;
    int tid = threadIdx.x;
    for (int i = tid; i < 49 * 32; i += 128) {
        int rr = i >> 5, dd = i & 31;
        size_t g = ((size_t)(win * 49 + rr)) * C_ + head * HD_ + dd;
        sq[rr][dd] = qf[g]; sk[rr][dd] = kf[g]; sv[rr][dd] = vf[g];
        sa1[rr][dd] = a1f[g]; sa2[rr][dd] = a2f[g];
    }
    __syncthreads();
    const float* pbh = pb + head * 2401;
    for (int i = tid; i < 2401; i += 128) {
        int a = i / 49, n = i % 49;
        float acc = 0.f;
#pragma unroll
        for (int d = 0; d < 32; ++d) acc += sa2[a][d] * sk[n][d];
        S[a][n] = acc * SCALE_ + pbh[i];
    }
    __syncthreads();
    if (tid < 49) {
        float mx = -1e30f;
        for (int n = 0; n < 49; ++n) mx = fmaxf(mx, S[tid][n]);
        float sum = 0.f;
        for (int n = 0; n < 49; ++n) { float e = __expf(S[tid][n] - mx); S[tid][n] = e; sum += e; }
        float inv = 1.f / sum;
        for (int n = 0; n < 49; ++n) S[tid][n] *= inv;
    }
    __syncthreads();
    for (int i = tid; i < 49 * 32; i += 128) {
        int a = i >> 5, dd = i & 31;
        float acc = 0.f;
        for (int n = 0; n < 49; ++n) acc += S[a][n] * sv[n][dd];
        av[a][dd] = acc;
    }
    __syncthreads();
    const float* abh = ab + head * 2401;
    for (int i = tid; i < 2401; i += 128) {
        int n = i / 49, a = i % 49;
        float acc = 0.f;
#pragma unroll
        for (int d = 0; d < 32; ++d) acc += sq[n][d] * sa1[a][d];
        S[n][a] = acc * SCALE_ + abh[i];
    }
    __syncthreads();
    if (tid < 49) {
        float mx = -1e30f;
        for (int a = 0; a < 49; ++a) mx = fmaxf(mx, S[tid][a]);
        float sum = 0.f;
        for (int a = 0; a < 49; ++a) { float e = __expf(S[tid][a] - mx); S[tid][a] = e; sum += e; }
        float inv = 1.f / sum;
        for (int a = 0; a < 49; ++a) S[tid][a] *= inv;
    }
    __syncthreads();
    for (int i = tid; i < 49 * 32; i += 128) {
        int n = i >> 5, dd = i & 31;
        float acc = 0.f;
        for (int a = 0; a < 49; ++a) acc += S[n][a] * av[a][dd];
        xw[((size_t)(win * 49 + n)) * C_ + head * HD_ + dd] = acc;
    }
}

/* ---------------- depthwise 3x3 on q (per-window 7x7), fused into xw ---------------- */
__global__ void dwc_kernel(const float* __restrict__ qf, const float* __restrict__ wts,
                           const float* __restrict__ bias, float* __restrict__ xw,
                           __bf16* __restrict__ xw_bf) {
    size_t idx = (size_t)blockIdx.x * blockDim.x + threadIdx.x;
    if (idx >= (size_t)NTOK * C_) return;
    int c = (int)(idx % C_);
    size_t rest = idx / C_;
    int pos = (int)(rest % 49);
    size_t win = rest / 49;
    int iy = pos / 7, ix = pos % 7;
    float acc = bias[c];
    for (int dy = -1; dy <= 1; ++dy) {
        int y = iy + dy; if (y < 0 || y >= 7) continue;
        for (int dx = -1; dx <= 1; ++dx) {
            int x = ix + dx; if (x < 0 || x >= 7) continue;
            acc += qf[(win * 49 + y * 7 + x) * C_ + c] * wts[c * 9 + (dy + 1) * 3 + (dx + 1)];
        }
    }
    float val = xw[idx] + acc;
    xw[idx] = val;
    xw_bf[idx] = (__bf16)val;
}

/* ---------------- window reverse + residual + LN3 ---------------- */
__global__ __launch_bounds__(192)
void resid_ln3_kernel(const float* __restrict__ xF1, const float* __restrict__ proj,
                      const float* __restrict__ g3, const float* __restrict__ b3,
                      float* __restrict__ xres, float* __restrict__ xn3f,
                      __bf16* __restrict__ xn3h) {
    __shared__ float red[256];
    __shared__ float s_mean, s_rstd;
    int t = blockIdx.x, c = threadIdx.x;
    int b = t / 12544, n = t % 12544;
    int hh = n / 112, ww = n % 112;
    size_t m_row = (size_t)((((b * 16 + hh / 7) * 16 + ww / 7) * 49) + (hh % 7) * 7 + (ww % 7));
    size_t base = (size_t)t * C_ + c;
    float x = xF1[base] + proj[m_row * C_ + c];
    xres[base] = x;
    red[c] = x; if (c < 64) red[192 + c] = 0.f;
    __syncthreads();
    for (int s = 128; s > 0; s >>= 1) { if (c < s) red[c] += red[c + s]; __syncthreads(); }
    if (c == 0) s_mean = red[0] * (1.f / 192.f);
    __syncthreads();
    float m = s_mean, d = x - m;
    red[c] = d * d; if (c < 64) red[192 + c] = 0.f;
    __syncthreads();
    for (int s = 128; s > 0; s >>= 1) { if (c < s) red[c] += red[c + s]; __syncthreads(); }
    if (c == 0) s_rstd = rsqrtf(red[0] * (1.f / 192.f) + LNEPS);
    __syncthreads();
    float xn = d * s_rstd * g3[c] + b3[c];
    xn3f[base] = xn;
    xn3h[base] = (__bf16)xn;
}

/* ---------------- DFT along W (real -> 57 complex bins) ---------------- */
__global__ __launch_bounds__(192)
void dftw_kernel(const float* __restrict__ x, const float* __restrict__ tw,
                 float* __restrict__ Yr, float* __restrict__ Yi) {
    int c = threadIdx.x;
    int blk = blockIdx.x;               /* (b*112+h)*57 + kw */
    int kw = blk % KF_; int bh = blk / KF_;
    const float* row = x + (size_t)bh * 112 * C_ + c;
    float sr = 0.f, si = 0.f;
    for (int w = 0; w < 112; ++w) {
        int j = (w * kw) % 112;
        float xv = row[(size_t)w * C_];
        sr += xv * tw[j];
        si -= xv * tw[112 + j];
    }
    size_t o = ((size_t)bh * KF_ + kw) * C_ + c;
    Yr[o] = sr; Yi[o] = si;
}

/* ---------------- DFT along H (complex, 112 bins) ---------------- */
__global__ __launch_bounds__(192)
void dfth_kernel(const float* __restrict__ Ywr, const float* __restrict__ Ywi,
                 const float* __restrict__ tw, float* __restrict__ Yr, float* __restrict__ Yi) {
    int c = threadIdx.x;
    int blk = blockIdx.x;               /* (b*112+kh)*57 + kw */
    int kw = blk % KF_; int t = blk / KF_;
    int kh = t % 112, b = t / 112;
    float sr = 0.f, si = 0.f;
    for (int h = 0; h < 112; ++h) {
        int j = (h * kh) % 112;
        float cr = tw[j], ci = tw[112 + j];
        size_t g = (((size_t)b * 112 + h) * KF_ + kw) * C_ + c;
        float ar = Ywr[g], ai = Ywi[g];
        sr += ar * cr + ai * ci;        /* (ar+i*ai)*(cr - i*ci) */
        si += ai * cr - ar * ci;
    }
    size_t o = (size_t)blk * C_ + c;
    Yr[o] = sr; Yi[o] = si;
}

/* ---------------- inverse DFT along H ---------------- */
__global__ __launch_bounds__(192)
void idfth_kernel(const float* __restrict__ Y2r, const float* __restrict__ Y2i,
                  const float* __restrict__ tw, float* __restrict__ Zr, float* __restrict__ Zi) {
    int c = threadIdx.x;
    int blk = blockIdx.x;               /* (b*112+h)*57 + kw */
    int kw = blk % KF_; int t = blk / KF_;
    int h = t % 112, b = t / 112;
    float sr = 0.f, si = 0.f;
    for (int kh = 0; kh < 112; ++kh) {
        int j = (h * kh) % 112;
        float cr = tw[j], ci = tw[112 + j];
        size_t g = (((size_t)b * 112 + kh) * KF_ + kw) * C_ + c;
        float ar = Y2r[g], ai = Y2i[g];
        sr += ar * cr - ai * ci;        /* (ar+i*ai)*(cr + i*ci) */
        si += ar * ci + ai * cr;
    }
    size_t o = (size_t)blk * C_ + c;
    Zr[o] = sr * (1.f / 112.f);
    Zi[o] = si * (1.f / 112.f);
}

/* ---------------- inverse rDFT along W + final out = x + mlp + fft ---------------- */
__global__ __launch_bounds__(192)
void idftw_final_kernel(const float* __restrict__ Zr, const float* __restrict__ Zi,
                        const float* __restrict__ tw, const float* __restrict__ xres,
                        const float* __restrict__ mlp, float* __restrict__ out) {
    int c = threadIdx.x;
    int blk = blockIdx.x;               /* token index: b*12544 + h*112 + w */
    int w = blk % 112;
    size_t zbase = ((size_t)(blk / 112) * KF_) * C_ + c;
    float acc = Zr[zbase];              /* kw = 0 */
    for (int kw = 1; kw < 56; ++kw) {
        int j = (w * kw) % 112;
        size_t g = zbase + (size_t)kw * C_;
        acc += 2.f * (Zr[g] * tw[j] - Zi[g] * tw[112 + j]);
    }
    {
        int j = (w * 56) % 112;         /* Nyquist bin, counted once */
        size_t g = zbase + (size_t)56 * C_;
        acc += Zr[g] * tw[j] - Zi[g] * tw[112 + j];
    }
    acc *= (1.f / 112.f);
    size_t o = (size_t)blk * C_ + c;
    out[o] = xres[o] + mlp[o] + acc;
}

/* =================================================================== */
extern "C" void kernel_launch(void* const* d_in, const int* in_sizes, int n_in,
                              void* d_out, int out_size, void* d_ws, size_t ws_size,
                              hipStream_t stream) {
    (void)in_sizes; (void)n_in; (void)out_size; (void)ws_size;
    const float* x_F1   = (const float*)d_in[0];
    const float* x_C2   = (const float*)d_in[1];
    const float* x_C1   = (const float*)d_in[2];
    const float* g1     = (const float*)d_in[3];
    const float* b1     = (const float*)d_in[4];
    const float* Wq     = (const float*)d_in[5];
    const float* Wk     = (const float*)d_in[6];
    const float* Wv     = (const float*)d_in[7];
    const float* Wa1    = (const float*)d_in[8];
    const float* Wa2    = (const float*)d_in[9];
    const float* Wp     = (const float*)d_in[10];
    const float* bq     = (const float*)d_in[11];
    const float* bk     = (const float*)d_in[12];
    const float* bv     = (const float*)d_in[13];
    const float* ba1    = (const float*)d_in[14];
    const float* ba2    = (const float*)d_in[15];
    const float* bp     = (const float*)d_in[16];
    const float* dwc_w  = (const float*)d_in[17];
    const float* dwc_b  = (const float*)d_in[18];
    const float* an_bias= (const float*)d_in[19];
    const float* na_bias= (const float*)d_in[20];
    const float* ah_bias= (const float*)d_in[21];
    const float* aw_bias= (const float*)d_in[22];
    const float* ha_bias= (const float*)d_in[23];
    const float* wa_bias= (const float*)d_in[24];
    const float* g3     = (const float*)d_in[25];
    const float* b3     = (const float*)d_in[26];
    const float* Wfc1   = (const float*)d_in[27];
    const float* bfc1   = (const float*)d_in[28];
    const float* Wfc2   = (const float*)d_in[29];
    const float* bfc2   = (const float*)d_in[30];
    const float* fw1r   = (const float*)d_in[31];
    const float* fw1i   = (const float*)d_in[32];
    const float* fw2r   = (const float*)d_in[33];
    const float* fw2i   = (const float*)d_in[34];

    /* ---- carve workspace ---- */
    char* base = (char*)d_ws;
    size_t off = 0;
    auto take = [&](size_t bytes) -> void* {
        void* r = base + off;
        off += (bytes + 255) & ~(size_t)255;
        return r;
    };
    const size_t TC  = (size_t)NTOK * C_;        /* 19.27M elems */
    const size_t FC  = (size_t)NFREQ * C_;       /* 9.8M elems   */
    __bf16* qw_bf  = (__bf16*)take(TC * 2);
    __bf16* kin_bf = (__bf16*)take(TC * 2);
    __bf16* vin_bf = (__bf16*)take(TC * 2);
    __bf16* c1w_bf = (__bf16*)take(TC * 2);
    __bf16* c2w_bf = (__bf16*)take(TC * 2);
    float*  qf   = (float*)take(TC * 4);
    float*  kf   = (float*)take(TC * 4);
    float*  vf   = (float*)take(TC * 4);
    float*  a1f  = (float*)take(TC * 4);
    float*  a2f  = (float*)take(TC * 4);
    float*  xw   = (float*)take(TC * 4);
    __bf16* xw_bf= (__bf16*)take(TC * 2);
    float*  proj = (float*)take(TC * 4);
    float*  xres = (float*)take(TC * 4);
    float*  xn3f = (float*)take(TC * 4);
    __bf16* xn3h = (__bf16*)take(TC * 2);
    __bf16* hbf  = (__bf16*)take((size_t)NTOK * 768 * 2);
    float*  mlp  = (float*)take(TC * 4);
    float*  Ywr  = (float*)take(FC * 4);
    float*  Ywi  = (float*)take(FC * 4);
    float*  Yhr  = (float*)take(FC * 4);
    float*  Yhi  = (float*)take(FC * 4);
    float*  Gr   = (float*)take(FC * 4);
    float*  Gi   = (float*)take(FC * 4);
    float*  Y2r  = (float*)take(FC * 4);
    float*  Y2i  = (float*)take(FC * 4);
    float*  Zr   = (float*)take(FC * 4);
    float*  Zi   = (float*)take(FC * 4);
    __bf16* Wq_bt  = (__bf16*)take(36864 * 2);   /* all weight mats pre-transposed (NxK) */
    __bf16* Wk_bt  = (__bf16*)take(36864 * 2);
    __bf16* Wv_bt  = (__bf16*)take(36864 * 2);
    __bf16* Wa1_bt = (__bf16*)take(36864 * 2);
    __bf16* Wa2_bt = (__bf16*)take(36864 * 2);
    __bf16* Wp_bt  = (__bf16*)take(36864 * 2);
    __bf16* Wfc1_bt= (__bf16*)take(147456 * 2);
    __bf16* Wfc2_bt= (__bf16*)take(147456 * 2);
    float*  tw  = (float*)take(224 * 4);
    float*  pb  = (float*)take(14406 * 4);
    float*  ab  = (float*)take(14406 * 4);

    /* ---- weight conversion (transpose to N-major) + tables ---- */
    f2bft_kernel<<<(36864 + 255) / 256, 256, 0, stream>>>(Wq,  Wq_bt,  192, 192);
    f2bft_kernel<<<(36864 + 255) / 256, 256, 0, stream>>>(Wk,  Wk_bt,  192, 192);
    f2bft_kernel<<<(36864 + 255) / 256, 256, 0, stream>>>(Wv,  Wv_bt,  192, 192);
    f2bft_kernel<<<(36864 + 255) / 256, 256, 0, stream>>>(Wa1, Wa1_bt, 192, 192);
    f2bft_kernel<<<(36864 + 255) / 256, 256, 0, stream>>>(Wa2, Wa2_bt, 192, 192);
    f2bft_kernel<<<(36864 + 255) / 256, 256, 0, stream>>>(Wp,  Wp_bt,  192, 192);
    f2bft_kernel<<<(147456 + 255) / 256, 256, 0, stream>>>(Wfc1, Wfc1_bt, 192, 768);
    f2bft_kernel<<<(147456 + 255) / 256, 256, 0, stream>>>(Wfc2, Wfc2_bt, 768, 192);
    twiddle_init_kernel<<<1, 128, 0, stream>>>(tw);
    biastab_kernel<<<(14406 + 255) / 256, 256, 0, stream>>>(an_bias, na_bias, ah_bias,
                                                            aw_bias, ha_bias, wa_bias, pb, ab);

    /* ---- LN1 + window partition ---- */
    prep_kernel<<<NTOK, 192, 0, stream>>>(x_F1, x_C2, x_C1, g1, b1,
                                          qw_bf, kin_bf, vin_bf, c1w_bf, c2w_bf);

    /* ---- q,k,v,a1,a2 projections (bf16 WMMA, fully async-staged) ---- */
    dim3 g192(192 / GBN, NTOK / GBM);       /* (3, 784) */
    gemm_bf16_kernel<<<g192, 256, 0, stream>>>(qw_bf,  Wq_bt,  bq,  qf,  nullptr, NTOK, 192, 192, 0);
    gemm_bf16_kernel<<<g192, 256, 0, stream>>>(kin_bf, Wk_bt,  bk,  kf,  nullptr, NTOK, 192, 192, 0);
    gemm_bf16_kernel<<<g192, 256, 0, stream>>>(vin_bf, Wv_bt,  bv,  vf,  nullptr, NTOK, 192, 192, 0);
    gemm_bf16_kernel<<<g192, 256, 0, stream>>>(c1w_bf, Wa1_bt, ba1, a1f, nullptr, NTOK, 192, 192, 0);
    gemm_bf16_kernel<<<g192, 256, 0, stream>>>(c2w_bf, Wa2_bt, ba2, a2f, nullptr, NTOK, 192, 192, 0);

    /* ---- agent attention + DWC + proj ---- */
    attn_kernel<<<NWIN * HEADS_, 128, 0, stream>>>(qf, kf, vf, a1f, a2f, pb, ab, xw);
    dwc_kernel<<<(int)((TC + 255) / 256), 256, 0, stream>>>(qf, dwc_w, dwc_b, xw, xw_bf);
    gemm_bf16_kernel<<<g192, 256, 0, stream>>>(xw_bf, Wp_bt, bp, proj, nullptr, NTOK, 192, 192, 0);

    /* ---- residual + LN3 ---- */
    resid_ln3_kernel<<<NTOK, 192, 0, stream>>>(x_F1, proj, g3, b3, xres, xn3f, xn3h);

    /* ---- MLP (bf16 WMMA, exact GELU) ---- */
    dim3 g768(768 / GBN, NTOK / GBM);       /* (12, 784) */
    gemm_bf16_kernel<<<g768, 256, 0, stream>>>(xn3h, Wfc1_bt, bfc1, nullptr, hbf, NTOK, 768, 192, 1);
    gemm_bf16_kernel<<<g192, 256, 0, stream>>>(hbf,  Wfc2_bt, bfc2, mlp, nullptr, NTOK, 192, 768, 0);

    /* ---- spectral branch ---- */
    dftw_kernel<<<NFREQ, 192, 0, stream>>>(xn3f, tw, Ywr, Ywi);
    dfth_kernel<<<NFREQ, 192, 0, stream>>>(Ywr, Ywi, tw, Yhr, Yhi);
    dim3 gfr(192 / 64, NFREQ / 128);        /* (3, 399) */
    gemm_f32x2_kernel<<<gfr, 256, 0, stream>>>(Yhr, fw1r, Yhi, fw1i, -1.f, Gr,  NFREQ, 192, 192, 1);
    gemm_f32x2_kernel<<<gfr, 256, 0, stream>>>(Yhr, fw1i, Yhi, fw1r, +1.f, Gi,  NFREQ, 192, 192, 1);
    gemm_f32x2_kernel<<<gfr, 256, 0, stream>>>(Gr,  fw2r, Gi,  fw2i, -1.f, Y2r, NFREQ, 192, 192, 0);
    gemm_f32x2_kernel<<<gfr, 256, 0, stream>>>(Gr,  fw2i, Gi,  fw2r, +1.f, Y2i, NFREQ, 192, 192, 0);
    idfth_kernel<<<NFREQ, 192, 0, stream>>>(Y2r, Y2i, tw, Zr, Zi);

    /* ---- inverse rDFT along W fused with final sum ---- */
    idftw_final_kernel<<<NTOK, 192, 0, stream>>>(Zr, Zi, tw, xres, mlp, (float*)d_out);
}
